// KNNLoss_42417097015906
// MI455X (gfx1250) — compile-verified, matrix operand order checked by reading, not run on hardware
//
#include <hip/hip_runtime.h>
#include <hip/hip_bf16.h>

typedef __attribute__((ext_vector_type(16))) __bf16 v16bf;
typedef __attribute__((ext_vector_type(8)))  __bf16 v8bf;
typedef __attribute__((ext_vector_type(8)))  float  v8f;

// global (address_space(1)) pointer typedefs so the backend emits global_* not flat_*
typedef const __attribute__((address_space(1))) v8bf*  gcv8bf;
typedef const __attribute__((address_space(1))) float* gcf;
typedef __attribute__((address_space(1))) float*       gf;

#define GG    8
#define BB    4
#define FF    64
#define NN    8192
#define NPG   1024        // points per group
#define DD    192         // FF*3
#define NBATCH 32         // GG*BB
#define KSEL  16
#define CH    11
#define NROWS 32768       // NBATCH*NPG

// ---------------- kernel 1: group id per point (argmax of onehot) ----------------
__global__ void k_bg(const float* __restrict__ ex, const float* __restrict__ ac,
                     int* __restrict__ bgE, int* __restrict__ bgA) {
    int i = blockIdx.x * blockDim.x + threadIdx.x;
    if (i >= NN) return;
    const float* pe = ex + (size_t)i * CH + 3;   // b=0,f=0
    const float* pa = ac + (size_t)i * CH + 3;
    int be = 0, ba = 0; float me = pe[0], ma = pa[0];
    #pragma unroll
    for (int c = 1; c < 8; c++) {
        if (pe[c] > me) { me = pe[c]; be = c; }
        if (pa[c] > ma) { ma = pa[c]; ba = c; }
    }
    bgE[i] = be; bgA[i] = ba;
}

// ---------------- kernel 2: stable counting sort -> ord ----------------
__global__ void k_order(const int* __restrict__ bgE, const int* __restrict__ bgA,
                        int* __restrict__ ordE, int* __restrict__ ordA) {
    int t = threadIdx.x;
    if (t >= 16) return;
    const int* bg = (t < 8) ? bgE : bgA;
    int* ord      = (t < 8) ? ordE : ordA;
    int g = t & 7;
    int pos = 0;
    for (int i = 0; i < NN; i++) if (bg[i] < g) pos++;
    for (int i = 0; i < NN; i++) if (bg[i] == g) ord[pos++] = i;
}

// ---------------- kernel 3: per (g,b,f,c) sums and sumsq ----------------
__global__ void k_stats(const float* __restrict__ ex, const float* __restrict__ ac,
                        const int* __restrict__ ordE, const int* __restrict__ ordA,
                        float* __restrict__ S1e, float* __restrict__ S2e,
                        float* __restrict__ S1a, float* __restrict__ S2a) {
    int blk = blockIdx.x;              // ((g*BB+b)*FF + f)
    int f  = blk % FF;
    int gb = blk / FF;
    int b  = gb % BB;
    int g  = gb / BB;
    float acc[12];
    #pragma unroll
    for (int q = 0; q < 12; q++) acc[q] = 0.f;
    for (int j = threadIdx.x; j < NPG; j += blockDim.x) {
        int ie = ordE[g * NPG + j], ia = ordA[g * NPG + j];
        const float* pe = ex + (((size_t)b * FF + f) * NN + ie) * CH;
        const float* pa = ac + (((size_t)b * FF + f) * NN + ia) * CH;
        #pragma unroll
        for (int c = 0; c < 3; c++) {
            float ve = pe[c], va = pa[c];
            acc[c]     += ve; acc[3 + c] += ve * ve;
            acc[6 + c] += va; acc[9 + c] += va * va;
        }
    }
    __shared__ float red[128];
    for (int q = 0; q < 12; q++) {
        red[threadIdx.x] = acc[q];
        __syncthreads();
        for (int s = 64; s > 0; s >>= 1) {
            if ((int)threadIdx.x < s) red[threadIdx.x] += red[threadIdx.x + s];
            __syncthreads();
        }
        if (threadIdx.x == 0) {
            float* dst = (q < 3) ? S1e : (q < 6) ? S2e : (q < 9) ? S1a : S2a;
            dst[(size_t)gb * FF * 3 + f * 3 + (q % 3)] = red[0];
        }
        __syncthreads();
    }
}

// ---------------- kernel 4: means + inverse stds (ddof=1) ----------------
__global__ void k_norm_stats(const float* __restrict__ S1e, const float* __restrict__ S2e,
                             const float* __restrict__ S1a, const float* __restrict__ S2a,
                             float* __restrict__ meanE, float* __restrict__ meanA,
                             float* __restrict__ istdE, float* __restrict__ istdA) {
    int gb = blockIdx.x;
    int f  = threadIdx.x;   // 64 threads
    __shared__ float m[6];
    __shared__ float red[64];
    for (int q = 0; q < 6; q++) {
        const float* S1 = (q < 3) ? S1e : S1a;
        int c = q % 3;
        red[f] = S1[(size_t)gb * FF * 3 + f * 3 + c];
        __syncthreads();
        for (int s = 32; s > 0; s >>= 1) {
            if (f < s) red[f] += red[f + s];
            __syncthreads();
        }
        if (f == 0) m[q] = red[0] / (float)(FF * NPG);
        __syncthreads();
    }
    if (f == 0) {
        #pragma unroll
        for (int c = 0; c < 3; c++) { meanE[gb * 3 + c] = m[c]; meanA[gb * 3 + c] = m[3 + c]; }
    }
    for (int side = 0; side < 2; side++) {
        const float* S1 = side ? S1a : S1e;
        const float* S2 = side ? S2a : S2e;
        float sy = 0.f, sy2 = 0.f;
        #pragma unroll
        for (int c = 0; c < 3; c++) {
            float mc = m[side * 3 + c];
            float s1 = S1[(size_t)gb * FF * 3 + f * 3 + c];
            float s2 = S2[(size_t)gb * FF * 3 + f * 3 + c];
            sy  += s1 - (float)NPG * mc;
            sy2 += s2 - 2.f * mc * s1 + (float)NPG * mc * mc;
        }
        float mu  = sy / (3.f * NPG);
        float var = (sy2 - 3.f * NPG * mu * mu) / (3.f * NPG - 1.f);
        (side ? istdA : istdE)[gb * FF + f] = 1.0f / sqrtf(var);
    }
}

// ---------------- kernel 5: gather + velocity + normalize -> bf16 operands + row norms ----------------
__global__ void k_build(const float* __restrict__ ex, const float* __restrict__ ac,
                        const int* __restrict__ ordE, const int* __restrict__ ordA,
                        const float* __restrict__ meanE, const float* __restrict__ meanA,
                        const float* __restrict__ istdE, const float* __restrict__ istdA,
                        __bf16* __restrict__ Xe, __bf16* __restrict__ Ne, __bf16* __restrict__ Ve,
                        __bf16* __restrict__ Xa, __bf16* __restrict__ Na, __bf16* __restrict__ Va,
                        float* __restrict__ qXe, float* __restrict__ qNe, float* __restrict__ qVe,
                        float* __restrict__ qXa, float* __restrict__ qNa, float* __restrict__ qVa) {
    int blk  = blockIdx.x;
    int side = blk >> 15;          // 0 = expected, 1 = actual
    int row  = blk & (NROWS - 1);  // gb*NPG + j
    int gb = row / NPG;
    int j  = row - gb * NPG;
    int g  = gb / BB;
    int b  = gb % BB;
    const float* P    = side ? ac : ex;
    const int*   ord  = side ? ordA : ordE;
    const float* mean = side ? meanA : meanE;
    const float* istd = side ? istdA : istdE;
    __bf16* X  = side ? Xa : Xe;
    __bf16* Nm = side ? Na : Ne;
    __bf16* V  = side ? Va : Ve;
    float* qX = side ? qXa : qXe;
    float* qN = side ? qNa : qNe;
    float* qV = side ? qVa : qVe;

    int idx = ord[g * NPG + j];
    int d = threadIdx.x;           // 0..191
    int f = d / 3, c = d - 3 * f;
    size_t src = (((size_t)b * FF + f) * NN + idx) * CH + c;
    float p = P[src];
    float pv = 0.f;
    if (f > 0) pv = p - P[src - (size_t)NN * CH];
    float pn = (p - mean[gb * 3 + c]) * istd[gb * FF + f];
    __bf16 hx = (__bf16)p, hv = (__bf16)pv, hn = (__bf16)pn;
    size_t dst = (size_t)row * DD + d;
    X[dst] = hx; V[dst] = hv; Nm[dst] = hn;
    float fx = (float)hx, fv = (float)hv, fn = (float)hn;

    __shared__ float red[192];
    float sq[3] = { fx * fx, fn * fn, fv * fv };
    for (int q = 0; q < 3; q++) {
        red[d] = sq[q];
        __syncthreads();
        if (d < 96) red[d] += red[d + 96]; __syncthreads();
        if (d < 48) red[d] += red[d + 48]; __syncthreads();
        if (d < 24) red[d] += red[d + 24]; __syncthreads();
        if (d < 12) red[d] += red[d + 12]; __syncthreads();
        if (d < 6)  red[d] += red[d + 6];  __syncthreads();
        if (d < 3)  red[d] += red[d + 3];  __syncthreads();
        if (d == 0) {
            float tot = red[0] + red[1] + red[2];
            (q == 0 ? qX : q == 1 ? qN : qV)[row] = tot;
        }
        __syncthreads();
    }
}

// ---------------- WMMA fragment loader: 16 bf16 via two global_load_b128 ----------------
__device__ __forceinline__ v16bf load16g(const __bf16* p) {
    gcv8bf q = (gcv8bf)p;      // address_space(1): lowers to global_load_b128
    v8bf lo = q[0];            // K chunk at +0 bytes
    v8bf hi = q[2];            // K chunk at +32 bytes (+16 elements)
    v16bf r;
    #pragma unroll
    for (int e = 0; e < 8; e++) { r[e] = lo[e]; r[e + 8] = hi[e]; }
    return r;
}

// ---------------- kernel 6: WMMA distances + per-row top-K ----------------
__launch_bounds__(128)
__global__ void k_dist(const __bf16* __restrict__ Xe, const __bf16* __restrict__ Ne, const __bf16* __restrict__ Ve,
                       const __bf16* __restrict__ Xa, const __bf16* __restrict__ Na, const __bf16* __restrict__ Va,
                       const float* __restrict__ qXe, const float* __restrict__ qNe, const float* __restrict__ qVe,
                       const float* __restrict__ qXa, const float* __restrict__ qNa, const float* __restrict__ qVa,
                       float* __restrict__ partials) {
    extern __shared__ float smem[];               // 3 * 16 * 1024 floats = 192 KB
    float* dG = smem;
    float* dN = smem + 16 * 1024;
    float* dV = smem + 32 * 1024;

    int wg   = blockIdx.x;        // 0..2047
    int tile = wg & 63;           // 16-row tile within batch
    int gb   = wg >> 6;           // batch (g*BB+b)
    int lane = threadIdx.x & 31;
    int wave = threadIdx.x >> 5;  // 0..3
    int half = lane >> 4;
    int lrow = lane & 15;
    int row0 = tile * 16;

    // fully unrolled so all pointers (incl. LDS dest) are compile-time selected:
    // LDS writes become ds_store, operand loads become global_load
    #pragma unroll
    for (int m = 0; m < 3; m++) {
        const __bf16* Am  = (m == 0) ? Xe  : (m == 1) ? Ne  : Ve;
        const __bf16* Bm  = (m == 0) ? Xa  : (m == 1) ? Na  : Va;
        const float*  qam = (m == 0) ? qXe : (m == 1) ? qNe : qVe;
        const float*  qbm = (m == 0) ? qXa : (m == 1) ? qNa : qVa;
        float*        dst = (m == 0) ? dG  : (m == 1) ? dN  : dV;

        // A fragments: lane holds row M = lrow; K packing per ISA 16-bit A layout
        const __bf16* A = Am + ((size_t)gb * NPG + row0 + lrow) * DD;
        v16bf afrag[6];
        #pragma unroll
        for (int kk = 0; kk < 6; kk++) afrag[kk] = load16g(A + kk * 32 + 8 * half);

        // hoist this lane's 8 row-norms (M = r + 8*half) into registers
        float qrow[8];
        {
            gcf qp = (gcf)(qam + (size_t)gb * NPG + row0 + 8 * half);
            #pragma unroll
            for (int r = 0; r < 8; r++) qrow[r] = qp[r];
        }
        gcf qcolp = (gcf)(qbm + (size_t)gb * NPG);

        for (int ct = wave; ct < 64; ct += 4) {
            const __bf16* Bp = Bm + ((size_t)gb * NPG + ct * 16 + lrow) * DD;
            v8f acc = {};
            #pragma unroll
            for (int kk = 0; kk < 6; kk++) {
                v16bf bfrag = load16g(Bp + kk * 32 + 8 * half);
                acc = __builtin_amdgcn_wmma_f32_16x16x32_bf16(
                        false, afrag[kk], false, bfrag, (short)0, acc, false, false);
            }
            float qcol = qcolp[ct * 16 + lrow];
            #pragma unroll
            for (int r = 0; r < 8; r++) {
                float d2 = qrow[r] + qcol - 2.0f * acc[r];
                // raw v_sqrt_f32 (~1 ulp) — ample for a mean-of-distances output
                dst[(r + 8 * half) * 1024 + ct * 16 + lrow] =
                    __builtin_amdgcn_sqrtf(fmaxf(d2, 1e-12f));
            }
        }
    }
    __syncthreads();

    // per-row K-smallest selection (stable, matches jnp.argsort tie-breaking)
    __shared__ float red[48];
    if (threadIdx.x < 16) {
        int r = threadIdx.x;
        float best[KSEL]; int bidx[KSEL];
        #pragma unroll
        for (int k = 0; k < KSEL; k++) { best[k] = 3.4e38f; bidx[k] = 0; }
        const float* rowg = dG + r * 1024;
        for (int j = 0; j < 1024; j++) {
            float v = rowg[j];
            if (v < best[KSEL - 1]) {
                int p = KSEL - 1;
                while (p > 0 && v < best[p - 1]) { best[p] = best[p - 1]; bidx[p] = bidx[p - 1]; p--; }
                best[p] = v; bidx[p] = j;
            }
        }
        float sg = 0.f, sv = 0.f;
        #pragma unroll
        for (int k = 0; k < KSEL; k++) { sg += best[k]; sv += dV[r * 1024 + bidx[k]]; }
        float bn[KSEL];
        #pragma unroll
        for (int k = 0; k < KSEL; k++) bn[k] = 3.4e38f;
        const float* rown = dN + r * 1024;
        for (int j = 0; j < 1024; j++) {
            float v = rown[j];
            if (v < bn[KSEL - 1]) {
                int p = KSEL - 1;
                while (p > 0 && v < bn[p - 1]) { bn[p] = bn[p - 1]; p--; }
                bn[p] = v;
            }
        }
        float sn = 0.f;
        #pragma unroll
        for (int k = 0; k < KSEL; k++) sn += bn[k];
        red[r] = sg; red[16 + r] = sn; red[32 + r] = sv;
    }
    __syncthreads();
    if (threadIdx.x == 0) {
        float a = 0.f, b = 0.f, c = 0.f;
        for (int r = 0; r < 16; r++) { a += red[r]; b += red[16 + r]; c += red[32 + r]; }
        gf pp = (gf)partials;
        pp[wg * 3 + 0] = a;
        pp[wg * 3 + 1] = b;
        pp[wg * 3 + 2] = c;
    }
}

// ---------------- kernel 7: deterministic final reduction ----------------
__global__ void k_finish(const float* __restrict__ partials, float* __restrict__ out) {
    __shared__ float red[768];
    int t = threadIdx.x;   // 256 threads
    float a = 0.f, b = 0.f, c = 0.f;
    for (int i = t; i < 2048; i += 256) {
        a += partials[i * 3 + 0];
        b += partials[i * 3 + 1];
        c += partials[i * 3 + 2];
    }
    red[t] = a; red[256 + t] = b; red[512 + t] = c;
    __syncthreads();
    for (int s = 128; s > 0; s >>= 1) {
        if (t < s) {
            red[t]       += red[t + s];
            red[256 + t] += red[256 + t + s];
            red[512 + t] += red[512 + t + s];
        }
        __syncthreads();
    }
    if (t == 0) {
        const float scale = 1.0f / (524288.0f * 8.0f);   // mean over G*B*n*K, then / sqrt(F)
        out[0] = red[0]   * scale;
        out[1] = red[256] * scale;
        out[2] = red[512] * scale;
    }
}

extern "C" void kernel_launch(void* const* d_in, const int* in_sizes, int n_in,
                              void* d_out, int out_size, void* d_ws, size_t ws_size,
                              hipStream_t stream) {
    const float* ex = (const float*)d_in[0];
    const float* ac = (const float*)d_in[1];
    float* out = (float*)d_out;

    char* w = (char*)d_ws;
    size_t off = 0;
    auto alloc = [&](size_t bytes) -> char* {
        char* p = w + off;
        off += (bytes + 255) & ~(size_t)255;
        return p;
    };
    int* bgE  = (int*)alloc(NN * 4);
    int* bgA  = (int*)alloc(NN * 4);
    int* ordE = (int*)alloc(NN * 4);
    int* ordA = (int*)alloc(NN * 4);
    float* S1e = (float*)alloc((size_t)NBATCH * FF * 3 * 4);
    float* S2e = (float*)alloc((size_t)NBATCH * FF * 3 * 4);
    float* S1a = (float*)alloc((size_t)NBATCH * FF * 3 * 4);
    float* S2a = (float*)alloc((size_t)NBATCH * FF * 3 * 4);
    float* meanE = (float*)alloc(NBATCH * 3 * 4);
    float* meanA = (float*)alloc(NBATCH * 3 * 4);
    float* istdE = (float*)alloc(NBATCH * FF * 4);
    float* istdA = (float*)alloc(NBATCH * FF * 4);
    size_t matB = (size_t)NROWS * DD * sizeof(__bf16);
    __bf16* Xe = (__bf16*)alloc(matB);
    __bf16* Ne = (__bf16*)alloc(matB);
    __bf16* Ve = (__bf16*)alloc(matB);
    __bf16* Xa = (__bf16*)alloc(matB);
    __bf16* Na = (__bf16*)alloc(matB);
    __bf16* Va = (__bf16*)alloc(matB);
    float* qXe = (float*)alloc(NROWS * 4);
    float* qNe = (float*)alloc(NROWS * 4);
    float* qVe = (float*)alloc(NROWS * 4);
    float* qXa = (float*)alloc(NROWS * 4);
    float* qNa = (float*)alloc(NROWS * 4);
    float* qVa = (float*)alloc(NROWS * 4);
    float* partials = (float*)alloc(2048 * 3 * 4);

    k_bg<<<(NN + 255) / 256, 256, 0, stream>>>(ex, ac, bgE, bgA);
    k_order<<<1, 32, 0, stream>>>(bgE, bgA, ordE, ordA);
    k_stats<<<NBATCH * FF, 128, 0, stream>>>(ex, ac, ordE, ordA, S1e, S2e, S1a, S2a);
    k_norm_stats<<<NBATCH, 64, 0, stream>>>(S1e, S2e, S1a, S2a, meanE, meanA, istdE, istdA);
    k_build<<<NROWS * 2, 192, 0, stream>>>(ex, ac, ordE, ordA, meanE, meanA, istdE, istdA,
                                           Xe, Ne, Ve, Xa, Na, Va,
                                           qXe, qNe, qVe, qXa, qNa, qVa);
    k_dist<<<2048, 128, 3 * 16 * 1024 * sizeof(float), stream>>>(
        Xe, Ne, Ve, Xa, Na, Va, qXe, qNe, qVe, qXa, qNa, qVa, partials);
    k_finish<<<1, 256, 0, stream>>>(partials, out);
}
